// VectorQuantizer_82703890252202
// MI455X (gfx1250) — matrix operand order repaired and checked by reading, not compiled, not run on hardware
//
#include <hip/hip_runtime.h>

#define NTOK  131072
#define DIM   256
#define NCODE 1024
#define TILE_CODES 64
#define LDS_STRIDE 264   // 256 halfs + 8 pad: row stride 528B -> banks advance by 4, no 16-way conflict

typedef __attribute__((ext_vector_type(16))) _Float16 v16h;
typedef __attribute__((ext_vector_type(8)))  _Float16 v8h;
typedef __attribute__((ext_vector_type(8)))  float    v8f;

// CDNA5 async copy global->LDS (16B per lane, ASYNCcnt-tracked).
// INST_OFFSET is added to BOTH the global and the LDS address (ISA 15.18.3).
#define ASYNC_CP16(ldsoff, gaddr, imm)                                    \
    asm volatile("global_load_async_to_lds_b128 %0, %1, off offset:" #imm \
                 :: "v"(ldsoff), "v"(gaddr) : "memory")

// ---------------------------------------------------------------------------
// Kernel A: codebook f32 -> f16, and 0.5*||c_k||^2 (deterministic tree reduce)
// ---------------------------------------------------------------------------
__global__ void vq_prep(const float* __restrict__ cb, _Float16* __restrict__ cb16,
                        float* __restrict__ cnorm) {
    const int k = blockIdx.x;
    const int d = threadIdx.x;
    const float v = cb[(size_t)k * DIM + d];
    cb16[(size_t)k * DIM + d] = (_Float16)v;
    __shared__ float red[256];
    red[d] = v * v;
    __syncthreads();
    for (int s = 128; s > 0; s >>= 1) {
        if (d < s) red[d] += red[d + s];
        __syncthreads();
    }
    if (d == 0) cnorm[k] = 0.5f * red[0];
}

// ---------------------------------------------------------------------------
// Kernel B: WMMA distance GEMM + running argmax of (z.c - 0.5||c||^2).
// 8 waves/block, one 16-token tile per wave; codebook staged through LDS
// with GLOBAL_LOAD_ASYNC_TO_LDS_B128; B fragments software-pipelined so
// ds_load latency overlaps the WMMA chain.
// ---------------------------------------------------------------------------
__global__ __launch_bounds__(256) void vq_argmin(
    const float* __restrict__ z, const _Float16* __restrict__ cb16,
    const float* __restrict__ cnorm, int* __restrict__ bestIdx,
    float* __restrict__ outIdxF) {
    __shared__ _Float16 tile[TILE_CODES * LDS_STRIDE];

    const int tid  = threadIdx.x;
    const int lane = tid & 31;
    const int wave = tid >> 5;
    const int nl   = lane & 15;   // row (A) / column (B,D) slot within 16x16 tile
    const int hi   = lane >> 4;   // half-wave select
    const int rowBase = blockIdx.x * 128 + wave * 16;

    // Async-staging addresses: 256 threads x 128B cover 64 codes x 512B.
    const int r = tid >> 2;       // code row in tile (0..63)
    const int q = tid & 3;        // quarter of the row (64 halfs)
    const unsigned ldsDst =
        (unsigned)(size_t)(tile + r * LDS_STRIDE + q * 64);  // low 32 bits = LDS offset

    // Preload A: 16 tokens x 256 dims as 8 f16 16x32 fragments (contiguous k order).
    // lanes 0-15: halves 0-15 -> K=f*32+0..15
    // lanes16-31: halves 0-15 -> K=f*32+16..31
    v16h A[8];
    const float* zrow = z + (size_t)(rowBase + nl) * DIM;
    #pragma unroll
    for (int f = 0; f < 8; ++f) {
        const int base = f * 32 + hi * 16;
        #pragma unroll
        for (int h = 0; h < 8; ++h) {
            A[f][h]     = (_Float16)zrow[base + h];
            A[f][h + 8] = (_Float16)zrow[base + 8 + h];
        }
    }

    float bestV[8];
    int   bestI[8];
    #pragma unroll
    for (int v = 0; v < 8; ++v) { bestV[v] = -3.4e38f; bestI[v] = 0; }

    for (int cb0 = 0; cb0 < NCODE; cb0 += TILE_CODES) {
        __syncthreads();   // previous tile fully consumed
        {
            const _Float16* src = cb16 + (size_t)(cb0 + r) * DIM + q * 64;
            const unsigned long long g = (unsigned long long)(size_t)src;
            ASYNC_CP16(ldsDst, g, 0);
            ASYNC_CP16(ldsDst, g, 16);
            ASYNC_CP16(ldsDst, g, 32);
            ASYNC_CP16(ldsDst, g, 48);
            ASYNC_CP16(ldsDst, g, 64);
            ASYNC_CP16(ldsDst, g, 80);
            ASYNC_CP16(ldsDst, g, 96);
            ASYNC_CP16(ldsDst, g, 112);
            asm volatile("s_wait_asynccnt 0x0" ::: "memory");
        }
        __syncthreads();   // all waves' async writes visible

        #pragma unroll
        for (int sub = 0; sub < 4; ++sub) {
            const int   n  = cb0 + sub * 16 + nl;
            const float cn = cnorm[n];          // hoisted: overlaps the WMMAs
            v8f acc[2];
            acc[0] = (v8f){0.f,0.f,0.f,0.f,0.f,0.f,0.f,0.f};
            acc[1] = (v8f){0.f,0.f,0.f,0.f,0.f,0.f,0.f,0.f};
            // B layout (32x16 f16): lane n=nl; lanes 0-15 hold K=ks*32+0..15,
            // lanes 16-31 hold K=ks*32+16..31, halves in ascending K order.
            const _Float16* brow = tile + (sub * 16 + nl) * LDS_STRIDE + hi * 16;
            // One-deep software pipeline: load(ks+1) issued before wmma(ks).
            v8h nb0 = *(const v8h*)(brow);
            v8h nb1 = *(const v8h*)(brow + 8);
            #pragma unroll
            for (int ks = 0; ks < 8; ++ks) {    // two independent WMMA chains
                v16h B;
                #pragma unroll
                for (int h = 0; h < 8; ++h) { B[h] = nb0[h]; B[h + 8] = nb1[h]; }
                if (ks < 7) {
                    nb0 = *(const v8h*)(brow + (ks + 1) * 32);
                    nb1 = *(const v8h*)(brow + (ks + 1) * 32 + 8);
                }
                acc[ks & 1] = __builtin_amdgcn_wmma_f32_16x16x32_f16(
                    false, A[ks], false, B, (short)0, acc[ks & 1], false, false);
            }
            #pragma unroll
            for (int v = 0; v < 8; ++v) {
                const float s = (acc[0][v] + acc[1][v]) - cn;  // max z.c - 0.5||c||^2
                if (s > bestV[v]) { bestV[v] = s; bestI[v] = n; }
            }
        }
    }

    // Argmax across the 16 candidate columns (half-wave butterfly).
    // Tie-break: smaller code index wins (matches jnp.argmin first-min).
    #pragma unroll
    for (int v = 0; v < 8; ++v) {
        #pragma unroll
        for (int mask = 1; mask <= 8; mask <<= 1) {
            const float ov = __shfl_xor(bestV[v], mask, 32);
            const int   oi = __shfl_xor(bestI[v], mask, 32);
            if (ov > bestV[v] || (ov == bestV[v] && oi < bestI[v])) {
                bestV[v] = ov; bestI[v] = oi;
            }
        }
    }
    // D layout: lanes 0-15 own rows 0-7 (VGPR v -> M=v), lanes 16-31 rows 8-15.
    if (lane == 0) {
        #pragma unroll
        for (int v = 0; v < 8; ++v) {
            const int row = rowBase + v;
            bestIdx[row] = bestI[v];
            outIdxF[row] = (float)bestI[v];
        }
    } else if (lane == 16) {
        #pragma unroll
        for (int v = 0; v < 8; ++v) {
            const int row = rowBase + 8 + v;
            bestIdx[row] = bestI[v];
            outIdxF[row] = (float)bestI[v];
        }
    }
}

// ---------------------------------------------------------------------------
// Kernel C: one wave per token. Gather z_q (f32 originals), emit
// z_st = z + (z_q - z), per-token squared error via shfl butterfly
// (fixed order -> deterministic across graph replays).
// ---------------------------------------------------------------------------
__global__ __launch_bounds__(256) void vq_gather(
    const float* __restrict__ z, const float* __restrict__ cb,
    const int* __restrict__ bestIdx, float* __restrict__ zst,
    float* __restrict__ tokErr) {
    const int wave = threadIdx.x >> 5;
    const int lane = threadIdx.x & 31;
    const int t = blockIdx.x * 8 + wave;
    const int idx = bestIdx[t];
    const float4* zr   = (const float4*)(z  + (size_t)t   * DIM);
    const float4* cr   = (const float4*)(cb + (size_t)idx * DIM);
    float4*       orow = (float4*)(zst + (size_t)t * DIM);
    float s = 0.f;
    #pragma unroll
    for (int i = 0; i < 2; ++i) {
        const float4 zv = zr[lane + 32 * i];
        const float4 cq = cr[lane + 32 * i];
        float4 o;
        o.x = zv.x + (cq.x - zv.x);   // mimic reference rounding of z + (z_q - z)
        o.y = zv.y + (cq.y - zv.y);
        o.z = zv.z + (cq.z - zv.z);
        o.w = zv.w + (cq.w - zv.w);
        orow[lane + 32 * i] = o;
        const float dx = zv.x - cq.x, dy = zv.y - cq.y;
        const float dz = zv.z - cq.z, dw = zv.w - cq.w;
        s += dx * dx + dy * dy + dz * dz + dw * dw;
    }
    #pragma unroll
    for (int mask = 16; mask >= 1; mask >>= 1) s += __shfl_xor(s, mask, 32);
    if (lane == 0) tokErr[t] = s;
}

// ---------------------------------------------------------------------------
// Kernel D: final loss = (1 + beta) * mean = 1.25 * sum / (N*D)
// ---------------------------------------------------------------------------
__global__ void vq_loss(const float* __restrict__ tokErr, float* __restrict__ outLoss) {
    __shared__ float red[256];
    const int d = threadIdx.x;
    float s = 0.f;
    for (int i = d; i < NTOK; i += 256) s += tokErr[i];
    red[d] = s;
    __syncthreads();
    for (int k = 128; k > 0; k >>= 1) {
        if (d < k) red[d] += red[d + k];
        __syncthreads();
    }
    if (d == 0) outLoss[0] = 1.25f * red[0] / ((float)NTOK * (float)DIM);
}

// ---------------------------------------------------------------------------
extern "C" void kernel_launch(void* const* d_in, const int* in_sizes, int n_in,
                              void* d_out, int out_size, void* d_ws, size_t ws_size,
                              hipStream_t stream) {
    (void)in_sizes; (void)n_in; (void)out_size; (void)ws_size;
    const float* z  = (const float*)d_in[0];
    const float* cb = (const float*)d_in[1];

    float* out      = (float*)d_out;
    float* zst      = out;                             // [NTOK*DIM]
    float* outIdxF  = out + (size_t)NTOK * DIM;        // [NTOK]
    float* outLoss  = outIdxF + NTOK;                  // [1]

    char* ws = (char*)d_ws;
    _Float16* cb16    = (_Float16*)ws;                                  // 512 KB
    float*    cnorm   = (float*)(ws + 524288);                          // 4 KB
    int*      bestIdx = (int*)(ws + 524288 + 4096);                     // 512 KB
    float*    tokErr  = (float*)(ws + 524288 + 4096 + (size_t)NTOK*4);  // 512 KB

    vq_prep  <<<NCODE,      256, 0, stream>>>(cb, cb16, cnorm);
    vq_argmin<<<NTOK / 128, 256, 0, stream>>>(z, cb16, cnorm, bestIdx, outIdxF);
    vq_gather<<<NTOK / 8,   256, 0, stream>>>(z, cb, bestIdx, zst, tokErr);
    vq_loss  <<<1,          256, 0, stream>>>(tokErr, outLoss);
}
